// MultiHeadedAttention_89773406421344
// MI455X (gfx1250) — compile-verified, matrix operand order checked by reading, not run on hardware
//
#include <hip/hip_runtime.h>

#define HEADS 16
#define DMODEL 1024
#define DK 64
#define SEQ 2048
#define BATCH 2
#define MROWS (SEQ * BATCH) /* 4096 */

typedef __attribute__((ext_vector_type(16))) __bf16 v16bf;
typedef __attribute__((ext_vector_type(8)))  float  v8f;
typedef unsigned short ushort_t;

union BF16Frag {
    v16bf v;
    ushort_t u[16];
    uint4 q[2];
};

__device__ __forceinline__ ushort_t f2bf(float f) {
    unsigned int x = __float_as_uint(f);
    unsigned int r = 0x7FFFu + ((x >> 16) & 1u);   // round-to-nearest-even
    return (ushort_t)((x + r) >> 16);
}

__device__ __forceinline__ v8f bf16_wmma(const BF16Frag& a, const BF16Frag& b, v8f c) {
    return __builtin_amdgcn_wmma_f32_16x16x32_bf16(
        /*neg_a=*/false, a.v, /*neg_b=*/false, b.v,
        /*c_mod=*/(short)0, c, /*reuse_a=*/false, /*reuse_b=*/false);
}

// Load a 16x32 bf16 fragment whose per-lane K window is
// {kb..kb+7, kb+16..kb+23}; p points at row_base + kb (bf16 elements).
__device__ __forceinline__ void frag_load(BF16Frag& f, const ushort_t* p) {
    f.q[0] = *(const uint4*)(p);
    f.q[1] = *(const uint4*)(p + 16);
}

// ---------------------------------------------------------------------------
// One-time fp32 -> bf16 conversion (4 elements / thread, n % 1024 == 0)
// ---------------------------------------------------------------------------
__global__ void __launch_bounds__(256)
cvt_f32_bf16(const float* __restrict__ in, ushort_t* __restrict__ out, int n) {
    const int i = (blockIdx.x * 256 + threadIdx.x) * 4;
    if (i >= n) return;
    const float4 f = *(const float4*)(in + i);
    union { ushort_t s[4]; uint2 u; } o;
    o.s[0] = f2bf(f.x); o.s[1] = f2bf(f.y);
    o.s[2] = f2bf(f.z); o.s[3] = f2bf(f.w);
    *(uint2*)(out + i) = o.u;
}

// ---------------------------------------------------------------------------
// Out = (X[M,K] @ W[N,K]^T + bias) * outScale      (bf16 operands, f32 accum)
// mode 0: store fp32 [M,N]
// mode 1: store bf16 [M,N]
// mode 2: store bf16 transposed V layout [b][h][dk][s]  (M row = s*BATCH+b)
// 32x32 tile per wave (2x2 register blocking): per K-step 8x b128 loads feed
// 4 WMMAs -> 2 loads/WMMA instead of 4, doubling XDL-feed arithmetic intensity.
// ---------------------------------------------------------------------------
__global__ void __launch_bounds__(256)
gemm_bf16_wmma(const ushort_t* __restrict__ X,
               const ushort_t* __restrict__ W,
               const float* __restrict__ bias,
               void* __restrict__ OutP,
               int M, int N, int K, int mode, float outScale) {
    const int lane = threadIdx.x & 31;
    const int wave = threadIdx.x >> 5;
    const int tile = blockIdx.x * 8 + wave;
    const int tilesN = N >> 5;
    const int tm = (tile / tilesN) << 5;
    const int tn = (tile % tilesN) << 5;
    const int l16 = lane & 15;
    const int hf  = lane >> 4;

    const ushort_t* xr0 = X + (size_t)(tm + l16) * K + hf * 8;
    const ushort_t* xr1 = xr0 + (size_t)16 * K;
    const ushort_t* wr0 = W + (size_t)(tn + l16) * K + hf * 8;
    const ushort_t* wr1 = wr0 + (size_t)16 * K;

    v8f acc00 = {}, acc01 = {}, acc10 = {}, acc11 = {};
#pragma unroll 2
    for (int k0 = 0; k0 < K; k0 += 32) {
        BF16Frag a0, a1, b0, b1;
        frag_load(a0, xr0 + k0);
        frag_load(a1, xr1 + k0);
        frag_load(b0, wr0 + k0);
        frag_load(b1, wr1 + k0);
        __builtin_prefetch(xr0 + k0 + 256, 0, 0);   // global_prefetch_b8
        __builtin_prefetch(xr1 + k0 + 256, 0, 0);
        __builtin_prefetch(wr0 + k0 + 256, 0, 0);
        __builtin_prefetch(wr1 + k0 + 256, 0, 0);
        acc00 = bf16_wmma(a0, b0, acc00);
        acc01 = bf16_wmma(a0, b1, acc01);
        acc10 = bf16_wmma(a1, b0, acc10);
        acc11 = bf16_wmma(a1, b1, acc11);
    }

    v8f* accs[4] = {&acc00, &acc01, &acc10, &acc11};
    if (mode == 0) {
        float* Out = (float*)OutP;
#pragma unroll
        for (int i = 0; i < 2; ++i)
#pragma unroll
            for (int j = 0; j < 2; ++j) {
                const v8f& a = *accs[i * 2 + j];
                const int n = tn + j * 16 + l16;
                const float bv = bias[n];
#pragma unroll
                for (int r = 0; r < 8; ++r) {
                    const int m = tm + i * 16 + hf * 8 + r;
                    Out[(size_t)m * N + n] = (a[r] + bv) * outScale;
                }
            }
    } else if (mode == 1) {
        ushort_t* Out = (ushort_t*)OutP;
#pragma unroll
        for (int i = 0; i < 2; ++i)
#pragma unroll
            for (int j = 0; j < 2; ++j) {
                const v8f& a = *accs[i * 2 + j];
                const int n = tn + j * 16 + l16;
                const float bv = bias[n];
#pragma unroll
                for (int r = 0; r < 8; ++r) {
                    const int m = tm + i * 16 + hf * 8 + r;
                    Out[(size_t)m * N + n] = f2bf((a[r] + bv) * outScale);
                }
            }
    } else {
        // transposed V store: [((b*HEADS + h)*DK + dk)*SEQ + s]
        ushort_t* Out = (ushort_t*)OutP;
#pragma unroll
        for (int i = 0; i < 2; ++i)
#pragma unroll
            for (int j = 0; j < 2; ++j) {
                const v8f& a = *accs[i * 2 + j];
                const int n  = tn + j * 16 + l16;
                const int h  = n / DK;
                const int dk = n % DK;
                const float bv = bias[n];
#pragma unroll
                for (int r = 0; r < 8; ++r) {
                    const int m = tm + i * 16 + hf * 8 + r;
                    const int s = m / BATCH;
                    const int b = m % BATCH;
                    Out[((size_t)(b * HEADS + h) * DK + dk) * SEQ + s] =
                        f2bf((a[r] + bv) * outScale);
                }
            }
    }
}

// ---------------------------------------------------------------------------
// Flash attention: one wave per (16-query tile, head, batch), keys streamed
// 32 at a time with online softmax.  Q/K bf16 in [s*B+b][h*64+dk] layout
// (Q pre-scaled by 1/8); V bf16 transposed [b][h][dk][s]; Ctx bf16 out.
// ---------------------------------------------------------------------------
__global__ void __launch_bounds__(32)
flash_attn_wmma(const ushort_t* __restrict__ Qb,
                const ushort_t* __restrict__ Kb,
                const ushort_t* __restrict__ Vt,
                ushort_t* __restrict__ Ctx) {
    __shared__ ushort_t sP[16 * 32];   // P tile staging (C-layout -> A-layout)

    const int lane = threadIdx.x;
    const int l16  = lane & 15;
    const int hf   = lane >> 4;
    const int qt   = blockIdx.x;
    const int h    = blockIdx.y >> 1;        // BATCH == 2
    const int b    = blockIdx.y & 1;

    BF16Frag aq0, aq1;
    {
        const ushort_t* qrow =
            Qb + (size_t)((qt * 16 + l16) * BATCH + b) * DMODEL + h * DK + hf * 8;
        frag_load(aq0, qrow);
        frag_load(aq1, qrow + 32);
    }

    v8f o[4] = {{}, {}, {}, {}};
    float mrun[8], lrun[8];
#pragma unroll
    for (int r = 0; r < 8; ++r) { mrun[r] = -1e30f; lrun[r] = 0.0f; }

    const ushort_t* vbase = Vt + (size_t)(b * HEADS + h) * DK * SEQ;

    for (int kt = 0; kt < SEQ / 32; ++kt) {
        // ---- scores: two 16-key groups, d_k=64 -> 2 WMMA K-steps each ----
        v8f s0 = {}, s1 = {};
        {
            const ushort_t* krow0 =
                Kb + (size_t)((kt * 32 + l16) * BATCH + b) * DMODEL + h * DK + hf * 8;
            const ushort_t* krow1 = krow0 + (size_t)16 * BATCH * DMODEL;
            BF16Frag bk;
            frag_load(bk, krow0);      s0 = bf16_wmma(aq0, bk, s0);
            frag_load(bk, krow0 + 32); s0 = bf16_wmma(aq1, bk, s0);
            frag_load(bk, krow1);      s1 = bf16_wmma(aq0, bk, s1);
            frag_load(bk, krow1 + 32); s1 = bf16_wmma(aq1, bk, s1);
        }

        // ---- online softmax (row stats within 16-lane groups) ----
        float p0[8], p1[8], alpha[8];
#pragma unroll
        for (int r = 0; r < 8; ++r) {
            float v = fmaxf(s0[r], s1[r]);
            v = fmaxf(v, __shfl_xor(v, 1, 32));
            v = fmaxf(v, __shfl_xor(v, 2, 32));
            v = fmaxf(v, __shfl_xor(v, 4, 32));
            v = fmaxf(v, __shfl_xor(v, 8, 32));
            const float mnew = fmaxf(mrun[r], v);
            p0[r] = __expf(s0[r] - mnew);
            p1[r] = __expf(s1[r] - mnew);
            float rs = p0[r] + p1[r];
            rs += __shfl_xor(rs, 1, 32);
            rs += __shfl_xor(rs, 2, 32);
            rs += __shfl_xor(rs, 4, 32);
            rs += __shfl_xor(rs, 8, 32);
            alpha[r] = __expf(mrun[r] - mnew);
            lrun[r]  = lrun[r] * alpha[r] + rs;
            mrun[r]  = mnew;
        }
#pragma unroll
        for (int t = 0; t < 4; ++t)
#pragma unroll
            for (int r = 0; r < 8; ++r) o[t][r] *= alpha[r];

        // ---- P: C-layout regs -> LDS -> A-layout fragment ----
#pragma unroll
        for (int r = 0; r < 8; ++r) {
            const int m = hf * 8 + r;
            sP[m * 32 + l16]      = f2bf(p0[r]);
            sP[m * 32 + 16 + l16] = f2bf(p1[r]);
        }
        __syncthreads();
        BF16Frag ap;
        frag_load(ap, &sP[l16 * 32 + hf * 8]);
        __syncthreads();

        // ---- P (16x32) x V (32x64): 4 WMMAs, B frags from transposed V ----
#pragma unroll
        for (int t = 0; t < 4; ++t) {
            BF16Frag bv;
            const ushort_t* vr =
                vbase + (size_t)(t * 16 + l16) * SEQ + kt * 32 + hf * 8;
            frag_load(bv, vr);
            o[t] = bf16_wmma(ap, bv, o[t]);
        }
    }

    // ---- normalize, store context as bf16 in [s*B+b][h*64+dk] ----
#pragma unroll
    for (int t = 0; t < 4; ++t)
#pragma unroll
        for (int r = 0; r < 8; ++r) {
            const int m = (qt * 16 + hf * 8 + r) * BATCH + b;
            Ctx[(size_t)m * DMODEL + h * DK + t * 16 + l16] =
                f2bf(o[t][r] / lrun[r]);
        }
}

extern "C" void kernel_launch(void* const* d_in, const int* in_sizes, int n_in,
                              void* d_out, int out_size, void* d_ws, size_t ws_size,
                              hipStream_t stream) {
    const float* query = (const float*)d_in[0];
    const float* key_  = (const float*)d_in[1];
    const float* value = (const float*)d_in[2];
    const float* Wq    = (const float*)d_in[3];
    const float* bq    = (const float*)d_in[4];
    const float* Wk    = (const float*)d_in[5];
    const float* bk    = (const float*)d_in[6];
    const float* Wv    = (const float*)d_in[7];
    const float* bv    = (const float*)d_in[8];
    const float* Wo    = (const float*)d_in[9];
    const float* bo    = (const float*)d_in[10];
    float* out = (float*)d_out;

    const size_t actN = (size_t)MROWS * DMODEL;   // 4 Mi elements
    const size_t wN   = (size_t)DMODEL * DMODEL;  // 1 Mi elements

    ushort_t* p   = (ushort_t*)d_ws;
    ushort_t* xqb = p;  p += actN;   // bf16 query
    ushort_t* xkb = p;  p += actN;   // bf16 key input
    ushort_t* xvb = p;  p += actN;   // bf16 value input
    ushort_t* wqb = p;  p += wN;     // bf16 weights
    ushort_t* wkb = p;  p += wN;
    ushort_t* wvb = p;  p += wN;
    ushort_t* wob = p;  p += wN;
    ushort_t* Qb  = p;  p += actN;   // bf16 Q (pre-scaled 1/8)
    ushort_t* Kb  = p;  p += actN;   // bf16 K
    ushort_t* Vt  = p;  p += actN;   // bf16 V transposed [b][h][dk][s]
    ushort_t* Ctx = p;  p += actN;   // bf16 attention output

    // ---- fp32 -> bf16 prep (one pass each) ----
    const int actBlocks = (int)(actN / 1024);   // 4 elems/thread, 256 threads
    const int wBlocks   = (int)(wN / 1024);
    cvt_f32_bf16<<<actBlocks, 256, 0, stream>>>(query, xqb, (int)actN);
    cvt_f32_bf16<<<actBlocks, 256, 0, stream>>>(key_,  xkb, (int)actN);
    cvt_f32_bf16<<<actBlocks, 256, 0, stream>>>(value, xvb, (int)actN);
    cvt_f32_bf16<<<wBlocks,   256, 0, stream>>>(Wq, wqb, (int)wN);
    cvt_f32_bf16<<<wBlocks,   256, 0, stream>>>(Wk, wkb, (int)wN);
    cvt_f32_bf16<<<wBlocks,   256, 0, stream>>>(Wv, wvb, (int)wN);
    cvt_f32_bf16<<<wBlocks,   256, 0, stream>>>(Wo, wob, (int)wN);

    // ---- projections: (M/32)*(N/32)=4096 tiles, 8 waves/block -> 512 blocks
    const dim3 ggrid(512), gblock(256);
    gemm_bf16_wmma<<<ggrid, gblock, 0, stream>>>(xqb, wqb, bq, Qb,
                                                 MROWS, DMODEL, DMODEL, 1, 0.125f);
    gemm_bf16_wmma<<<ggrid, gblock, 0, stream>>>(xkb, wkb, bk, Kb,
                                                 MROWS, DMODEL, DMODEL, 1, 1.0f);
    gemm_bf16_wmma<<<ggrid, gblock, 0, stream>>>(xvb, wvb, bv, Vt,
                                                 MROWS, DMODEL, DMODEL, 2, 1.0f);

    // ---- attention ----
    flash_attn_wmma<<<dim3(SEQ / 16, HEADS * BATCH), dim3(32), 0, stream>>>(
        Qb, Kb, Vt, Ctx);

    // ---- output projection (fp32 result) ----
    gemm_bf16_wmma<<<ggrid, gblock, 0, stream>>>(Ctx, wob, bo, out,
                                                 MROWS, DMODEL, DMODEL, 0, 1.0f);
}
